// MultiheadPeriodicAttention_79319456022757
// MI455X (gfx1250) — matrix operand order repaired and checked by reading, not compiled, not run on hardware
//
#include <hip/hip_runtime.h>
#include <hip/hip_bf16.h>
#include <stdint.h>

typedef __attribute__((ext_vector_type(16))) _Float16 v16h;
typedef __attribute__((ext_vector_type(8)))  _Float16 v8h;
typedef __attribute__((ext_vector_type(8)))  float    v8f;

#define B_  2
#define L_  2048
#define D_  1024
#define H_  16
#define DH_ 64
#define BH_ (B_*H_)
#define ML_ (B_*L_)   // 4096 rows of x

// ---------------------------------------------------------------------------
// CDNA5 async global->LDS copy (VGLOBAL async encoding, GV mode, ASYNCcnt).
// 16 bytes per lane; LDS dest address comes from a per-lane VGPR.
// ---------------------------------------------------------------------------
__device__ __forceinline__ void async_b128(void* lds_dst, const void* gsrc) {
  const uint32_t l = (uint32_t)(uintptr_t)lds_dst;   // low 32 bits = LDS offset
  const uint64_t g = (uint64_t)(uintptr_t)gsrc;
  asm volatile("global_load_async_to_lds_b128 %0, %1, off"
               :: "v"(l), "v"(g) : "memory");
}
__device__ __forceinline__ void wait_async0() {
  asm volatile("s_wait_asynccnt 0x0" ::: "memory");
}

// ---------------------------------------------------------------------------
// WMMA helpers (CDNA5 wave32, 16x16x32 f16 -> f32)
// ---------------------------------------------------------------------------
__device__ __forceinline__ v8f wmma_f16(v16h a, v16h b, v8f c) {
  return __builtin_amdgcn_wmma_f32_16x16x32_f16(false, a, false, b, (short)0, c,
                                                false, false);
}

// A fragment, 16x32 f16. LDS row-major [rows][stride].
// Lanes 0-15: row r, K {0..7,16..23}; lanes 16-31: row r, K {8..15,24..31}.
__device__ __forceinline__ v16h frag_a(const _Float16* lds, int row0, int stride,
                                       int k0, int lane) {
  const int r  = lane & 15;
  const int kb = (lane >> 4) << 3;           // 0 or 8
  const _Float16* p = lds + (size_t)(row0 + r) * stride + k0 + kb;
  v8h lo = *(const v8h*)(p);                 // K kb..kb+7
  v8h hi = *(const v8h*)(p + 16);            // K kb+16..kb+23
  v16h a;
#pragma unroll
  for (int i = 0; i < 8; ++i) { a[i] = lo[i]; a[i + 8] = hi[i]; }
  return a;
}

// B fragment, 32x16 f16, from transposed LDS tile [N][stride].
// Lanes 0-15: col c, K {0..7,16..23}; lanes 16-31: col c, K {8..15,24..31}.
__device__ __forceinline__ v16h frag_b(const _Float16* ldsT, int col0, int stride,
                                       int k0, int lane) {
  const int c  = lane & 15;
  const int kb = (lane >> 4) << 3;           // 0 or 8
  const _Float16* p = ldsT + (size_t)(col0 + c) * stride + k0 + kb;
  v8h lo = *(const v8h*)(p);
  v8h hi = *(const v8h*)(p + 16);
  v16h b;
#pragma unroll
  for (int i = 0; i < 8; ++i) { b[i] = lo[i]; b[i + 8] = hi[i]; }
  return b;
}

__device__ __forceinline__ void zero_acc(v8f* a, int n) {
  for (int i = 0; i < n; ++i)
#pragma unroll
    for (int e = 0; e < 8; ++e) a[i][e] = 0.0f;
}

// ---------------------------------------------------------------------------
// f32 -> f16 flat convert (x)
// ---------------------------------------------------------------------------
__global__ __launch_bounds__(256) void k_cvt(const float* __restrict__ src,
                                             _Float16* __restrict__ dst, int n) {
  int i = blockIdx.x * 256 + threadIdx.x;
  const int stride = gridDim.x * 256;
  for (; i < n; i += stride) dst[i] = (_Float16)src[i];
}

// ---------------------------------------------------------------------------
// Weight transpose + convert:  Wt[n][k] = (f16) W[k][n]   (D x D)
// ---------------------------------------------------------------------------
__global__ __launch_bounds__(256) void k_wtrans(
    const float* __restrict__ Wq, const float* __restrict__ Wk,
    const float* __restrict__ Wv, const float* __restrict__ Wo,
    _Float16* __restrict__ WqT, _Float16* __restrict__ WkT,
    _Float16* __restrict__ WvT, _Float16* __restrict__ WoT) {
  const float* S; _Float16* Dst;
  switch (blockIdx.z) {
    case 0: S = Wq; Dst = WqT; break;
    case 1: S = Wk; Dst = WkT; break;
    case 2: S = Wv; Dst = WvT; break;
    default: S = Wo; Dst = WoT; break;
  }
  __shared__ _Float16 tile[32][33];
  const int x  = blockIdx.x * 32 + threadIdx.x;  // n
  const int yb = blockIdx.y * 32;                // k base
  for (int r = threadIdx.y; r < 32; r += 8)
    tile[r][threadIdx.x] = (_Float16)S[(size_t)(yb + r) * D_ + x];
  __syncthreads();
  const int kk = yb + threadIdx.x;
  const int nb = blockIdx.x * 32;
  for (int r = threadIdx.y; r < 32; r += 8)
    Dst[(size_t)(nb + r) * D_ + kk] = tile[threadIdx.x][r];
}

// ---------------------------------------------------------------------------
// Fused QKV projection GEMM: [4096 x 1024] x [1024 x 1024] via WMMA.
// Double-buffered LDS, async global->LDS staging overlapped with WMMA.
// blockIdx.z: 0=q (pre-scaled, [bh][l][d]), 1=k ([bh][l][d]), 2=v ([bh][d][l]).
// ---------------------------------------------------------------------------
__global__ __launch_bounds__(256) void k_qkv_gemm(
    const _Float16* __restrict__ x16,
    const _Float16* __restrict__ WqT, const _Float16* __restrict__ WkT,
    const _Float16* __restrict__ WvT,
    const float* __restrict__ bq, const float* __restrict__ bk,
    const float* __restrict__ bv,
    _Float16* __restrict__ q16, _Float16* __restrict__ k16,
    _Float16* __restrict__ v16t) {
  constexpr int SA = 40;
  __shared__ _Float16 As[2][128 * SA];
  __shared__ _Float16 Bt[2][128 * SA];

  const int mode = blockIdx.z;
  const _Float16* WT = (mode == 0) ? WqT : (mode == 1) ? WkT : WvT;
  const float*    bb = (mode == 0) ? bq  : (mode == 1) ? bk  : bv;

  const int gn0 = blockIdx.x * 128;
  const int gm0 = blockIdx.y * 128;
  const int tid = threadIdx.x, lane = tid & 31, wid = tid >> 5;
  const int wm = (wid & 1) * 64;
  const int wn = (wid >> 1) * 32;

  v8f acc[8];
  zero_acc(acc, 8);

  const int row = tid >> 1;                    // 0..127
  const int cb  = (tid & 1) << 4;              // 0 or 16 (halves)
  const _Float16* gA = x16 + (size_t)(gm0 + row) * D_ + cb;
  const _Float16* gB = WT  + (size_t)(gn0 + row) * D_ + cb;

  auto issue = [&](int buf, int ko) {
    async_b128(&As[buf][row * SA + cb],     gA + ko);
    async_b128(&As[buf][row * SA + cb + 8], gA + ko + 8);
    async_b128(&Bt[buf][row * SA + cb],     gB + ko);
    async_b128(&Bt[buf][row * SA + cb + 8], gB + ko + 8);
  };

  constexpr int NIT = D_ / 32;
  issue(0, 0);
  for (int it = 0; it < NIT; ++it) {
    wait_async0();
    __syncthreads();
    if (it + 1 < NIT) issue((it + 1) & 1, (it + 1) * 32);

    const _Float16* a_lds = As[it & 1];
    const _Float16* b_lds = Bt[it & 1];
    v16h af[4], bf[2];
#pragma unroll
    for (int ti = 0; ti < 4; ++ti) af[ti] = frag_a(a_lds, wm + ti * 16, SA, 0, lane);
#pragma unroll
    for (int tj = 0; tj < 2; ++tj) bf[tj] = frag_b(b_lds, wn + tj * 16, SA, 0, lane);
#pragma unroll
    for (int ti = 0; ti < 4; ++ti)
#pragma unroll
      for (int tj = 0; tj < 2; ++tj)
        acc[ti * 2 + tj] = wmma_f16(af[ti], bf[tj], acc[ti * 2 + tj]);
  }

  const float scalef = (mode == 0) ? 0.125f : 1.0f;   // dh^-0.5 = 1/8
#pragma unroll
  for (int ti = 0; ti < 4; ++ti) {
#pragma unroll
    for (int tj = 0; tj < 2; ++tj) {
      const int n  = gn0 + wn + tj * 16 + (lane & 15);
      const int mb = gm0 + wm + ti * 16 + ((lane >> 4) << 3);
      const int h = n >> 6, d = n & 63;
      const float bias = bb[n];
#pragma unroll
      for (int i = 0; i < 8; ++i) {
        const int m = mb + i;
        const int b = m >> 11, l = m & (L_ - 1);
        const float val = (acc[ti * 2 + tj][i] + bias) * scalef;
        if (mode == 2) {
          v16t[((size_t)(b * H_ + h) * DH_ + d) * L_ + l] = (_Float16)val;
        } else {
          _Float16* dst = (mode == 0) ? q16 : k16;
          dst[((size_t)(b * H_ + h) * L_ + l) * DH_ + d] = (_Float16)val;
        }
      }
    }
  }
}

// ---------------------------------------------------------------------------
// Periodic bias: bias[b][j] = sum_{i=1..L-1} sin^2((t[b,i]-t[b,j]) * pi / p[b])
// ---------------------------------------------------------------------------
__global__ __launch_bounds__(256) void k_bias(const float* __restrict__ t,
                                              const float* __restrict__ p,
                                              float* __restrict__ biasv) {
  __shared__ float red[256];
  const int b = blockIdx.x >> 11;              // / L_
  const int j = blockIdx.x & (L_ - 1);
  const float tj  = t[(size_t)b * L_ + j];
  const float pip = 3.14159265358979323846f / p[b];
  float s = 0.f;
  for (int i = threadIdx.x; i < L_; i += 256) {
    if (i == 0) continue;                      // row i=0 forced to zero in ref
    const float sn = __sinf((t[(size_t)b * L_ + i] - tj) * pip);
    s += sn * sn;
  }
  red[threadIdx.x] = s;
  __syncthreads();
  for (int off = 128; off > 0; off >>= 1) {
    if (threadIdx.x < off) red[threadIdx.x] += red[threadIdx.x + off];
    __syncthreads();
  }
  if (threadIdx.x == 0) biasv[(size_t)b * L_ + j] = red[0];
}

// ---------------------------------------------------------------------------
// Scores: w_raw[bh][q][k] = q.k - bias[b][k], masked. 128x128 tile, K=dh=64.
// q/k tiles AND the 128x128 mask sub-block are async-staged to LDS up front
// (all waited on before the WMMAs); epilogue reads the mask from LDS with a
// conflict-free 144B padded stride instead of scattered global byte loads.
// ---------------------------------------------------------------------------
__global__ __launch_bounds__(256) void k_scores(
    const _Float16* __restrict__ q16, const _Float16* __restrict__ k16,
    const float* __restrict__ biasv, const uint8_t* __restrict__ mask,
    float* __restrict__ w) {
  constexpr int SA = 72;                       // 64 + 8 pad (halves)
  constexpr int SM = 144;                      // mask row stride (bytes)
  __shared__ _Float16 As[128 * SA];
  __shared__ _Float16 Bt[128 * SA];
  __shared__ uint8_t  Ml[128 * SM];            // mask tile [k_local][q_local]

  const int bh  = blockIdx.z;
  const int b   = bh >> 4;
  const int gn0 = blockIdx.x * 128;            // key tile
  const int gm0 = blockIdx.y * 128;            // query tile
  const int tid = threadIdx.x, lane = tid & 31, wid = tid >> 5;
  const int wm = (wid & 1) * 64, wn = (wid >> 1) * 32;

  const _Float16* qb = q16 + (size_t)bh * L_ * DH_;
  const _Float16* kb = k16 + (size_t)bh * L_ * DH_;

  {  // async stage 128x64 q tile, k tile, and 128x128 mask tile
    const int row = tid >> 1;
    const int cb  = (tid & 1) << 5;            // 0 or 32 halves
#pragma unroll
    for (int u = 0; u < 4; ++u) {
      async_b128(&As[row * SA + cb + u * 8], qb + (size_t)(gm0 + row) * DH_ + cb + u * 8);
      async_b128(&Bt[row * SA + cb + u * 8], kb + (size_t)(gn0 + row) * DH_ + cb + u * 8);
    }
    // mask[b][k][q]: row (gn0+row) of mask, 64 contiguous q-bytes per thread
    const int mcb = (tid & 1) << 6;            // 0 or 64 bytes
    const uint8_t* gm = mask + ((size_t)b * L_ + gn0 + row) * L_ + gm0 + mcb;
#pragma unroll
    for (int u = 0; u < 4; ++u)
      async_b128(&Ml[row * SM + mcb + u * 16], gm + u * 16);
  }
  wait_async0();
  __syncthreads();

  v8f acc[8];
  zero_acc(acc, 8);
#pragma unroll
  for (int k0 = 0; k0 < DH_; k0 += 32) {
    v16h af[4], bf[2];
#pragma unroll
    for (int ti = 0; ti < 4; ++ti) af[ti] = frag_a(As, wm + ti * 16, SA, k0, lane);
#pragma unroll
    for (int tj = 0; tj < 2; ++tj) bf[tj] = frag_b(Bt, wn + tj * 16, SA, k0, lane);
#pragma unroll
    for (int ti = 0; ti < 4; ++ti)
#pragma unroll
      for (int tj = 0; tj < 2; ++tj)
        acc[ti * 2 + tj] = wmma_f16(af[ti], bf[tj], acc[ti * 2 + tj]);
  }

#pragma unroll
  for (int ti = 0; ti < 4; ++ti) {
#pragma unroll
    for (int tj = 0; tj < 2; ++tj) {
      const int kl = wn + tj * 16 + (lane & 15);            // local key
      const int ql = wm + ti * 16 + ((lane >> 4) << 3);     // local query base
      const int kcol = gn0 + kl;
      const float kbias = biasv[(size_t)b * L_ + kcol];
#pragma unroll
      for (int i = 0; i < 8; ++i) {
        float val = acc[ti * 2 + tj][i] - kbias;
        if (!Ml[kl * SM + ql + i])                          // mask[b][k][q]
          val = -3.402823466e38f;
        w[((size_t)bh * L_ + gm0 + ql + i) * L_ + kcol] = val;
      }
    }
  }
}

// ---------------------------------------------------------------------------
// Row softmax over 2048 keys, in place on w (one block per row).
// ---------------------------------------------------------------------------
__global__ __launch_bounds__(256) void k_softmax(float* __restrict__ w) {
  __shared__ float red[256];
  float* base = w + (size_t)blockIdx.x * L_;
  const int tid = threadIdx.x;
  float v[8];
  float m = -3.402823466e38f;
#pragma unroll
  for (int j = 0; j < 8; ++j) {
    v[j] = base[tid + j * 256];
    m = fmaxf(m, v[j]);
  }
  red[tid] = m;
  __syncthreads();
  for (int off = 128; off > 0; off >>= 1) {
    if (tid < off) red[tid] = fmaxf(red[tid], red[tid + off]);
    __syncthreads();
  }
  m = red[0];
  __syncthreads();
  float s = 0.f;
#pragma unroll
  for (int j = 0; j < 8; ++j) {
    v[j] = __expf(v[j] - m);
    s += v[j];
  }
  red[tid] = s;
  __syncthreads();
  for (int off = 128; off > 0; off >>= 1) {
    if (tid < off) red[tid] += red[tid + off];
    __syncthreads();
  }
  const float inv = 1.0f / red[0];
#pragma unroll
  for (int j = 0; j < 8; ++j) base[tid + j * 256] = v[j] * inv;
}

// ---------------------------------------------------------------------------
// out_h = w @ v per (b,h): M=2048(q), N=64(d), K=2048(keys).
// B-side (v^T) async double-buffered; A-side (w, f32->f16) software-pipelined
// through registers so global loads overlap the WMMA block.
// ---------------------------------------------------------------------------
__global__ __launch_bounds__(256) void k_attnout(
    const float* __restrict__ w, const _Float16* __restrict__ v16t,
    _Float16* __restrict__ ao16) {
  constexpr int SA = 40;
  __shared__ _Float16 As[2][128 * SA];
  __shared__ _Float16 Bt[2][64 * SA];

  const int bh = blockIdx.z, b = bh >> 4, h = bh & 15;
  const int gm0 = blockIdx.x * 128;
  const int tid = threadIdx.x, lane = tid & 31, wid = tid >> 5;
  const int wm = (wid >> 1) * 32, wn = (wid & 1) * 32;

  const float*    wb = w + ((size_t)bh * L_ + gm0) * L_;
  const _Float16* vb = v16t + (size_t)bh * DH_ * L_;

  v8f acc[4];
  zero_acc(acc, 4);

  const int rowA = tid >> 1, cbA = (tid & 1) << 4;   // 16 f32 each
  const int rowB = tid >> 2, cbB = (tid & 3) << 3;   // 8 halves each
  const float*    gW = wb + (size_t)rowA * L_ + cbA;
  const _Float16* gV = vb + (size_t)rowB * L_ + cbB;

  auto loadA = [&](int ko, float4& f0, float4& f1, float4& f2, float4& f3) {
    const float4* g4 = (const float4*)(gW + ko);
    f0 = g4[0]; f1 = g4[1]; f2 = g4[2]; f3 = g4[3];
  };
  auto storeA = [&](int buf, float4 f0, float4 f1, float4 f2, float4 f3) {
    v8h h0, h1;
    h0[0] = (_Float16)f0.x; h0[1] = (_Float16)f0.y; h0[2] = (_Float16)f0.z; h0[3] = (_Float16)f0.w;
    h0[4] = (_Float16)f1.x; h0[5] = (_Float16)f1.y; h0[6] = (_Float16)f1.z; h0[7] = (_Float16)f1.w;
    h1[0] = (_Float16)f2.x; h1[1] = (_Float16)f2.y; h1[2] = (_Float16)f2.z; h1[3] = (_Float16)f2.w;
    h1[4] = (_Float16)f3.x; h1[5] = (_Float16)f3.y; h1[6] = (_Float16)f3.z; h1[7] = (_Float16)f3.w;
    *(v8h*)(&As[buf][rowA * SA + cbA])     = h0;
    *(v8h*)(&As[buf][rowA * SA + cbA + 8]) = h1;
  };
  auto issueB = [&](int buf, int ko) {
    async_b128(&Bt[buf][rowB * SA + cbB], gV + ko);
  };

  constexpr int NIT = L_ / 32;
  {
    float4 f0, f1, f2, f3;
    loadA(0, f0, f1, f2, f3);
    issueB(0, 0);
    storeA(0, f0, f1, f2, f3);
  }
  for (int it = 0; it < NIT; ++it) {
    wait_async0();
    __syncthreads();
    const bool nxt = (it + 1 < NIT);
    float4 n0, n1, n2, n3;
    if (nxt) {
      issueB((it + 1) & 1, (it + 1) * 32);
      loadA((it + 1) * 32, n0, n1, n2, n3);     // in flight during WMMAs
    }

    const _Float16* a_lds = As[it & 1];
    const _Float16* b_lds = Bt[it & 1];
    v16h af[2], bf[2];
#pragma unroll
    for (int ti = 0; ti < 2; ++ti) af[ti] = frag_a(a_lds, wm + ti * 16, SA, 0, lane);
#pragma unroll
    for (int tj = 0; tj < 2; ++tj) bf[tj] = frag_b(b_lds, wn + tj * 16, SA, 0, lane);
#pragma unroll
    for (int ti = 0; ti < 2; ++ti)
#pragma unroll
      for (int tj = 0; tj < 2; ++tj)
        acc[ti * 2 + tj] = wmma_f16(af[ti], bf[tj], acc[ti * 2 + tj]);

    if (nxt) storeA((it + 1) & 1, n0, n1, n2, n3);
  }

#pragma unroll
  for (int ti = 0; ti < 2; ++ti) {
#pragma unroll
    for (int tj = 0; tj < 2; ++tj) {
      const int d  = wn + tj * 16 + (lane & 15);
      const int mb = gm0 + wm + ti * 16 + ((lane >> 4) << 3);
#pragma unroll
      for (int i = 0; i < 8; ++i)
        ao16[((size_t)b * L_ + mb + i) * D_ + h * DH_ + d] =
            (_Float16)acc[ti * 2 + tj][i];
    }
  }
}

// ---------------------------------------------------------------------------
// Output projection: out = ao16 @ Wo + bo, f32 result to d_out.
// Same double-buffered async pipeline as k_qkv_gemm.
// ---------------------------------------------------------------------------
__global__ __launch_bounds__(256) void k_outproj(
    const _Float16* __restrict__ ao16, const _Float16* __restrict__ WoT,
    const float* __restrict__ bo, float* __restrict__ out) {
  constexpr int SA = 40;
  __shared__ _Float16 As[2][128 * SA];
  __shared__ _Float16 Bt[2][128 * SA];

  const int gn0 = blockIdx.x * 128;
  const int gm0 = blockIdx.y * 128;
  const int tid = threadIdx.x, lane = tid & 31, wid = tid >> 5;
  const int wm = (wid & 1) * 64, wn = (wid >> 1) * 32;

  v8f acc[8];
  zero_acc(acc, 8);

  const int row = tid >> 1, cb = (tid & 1) << 4;
  const _Float16* gA = ao16 + (size_t)(gm0 + row) * D_ + cb;
  const _Float16* gB = WoT  + (size_t)(gn0 + row) * D_ + cb;

  auto issue = [&](int buf, int ko) {
    async_b128(&As[buf][row * SA + cb],     gA + ko);
    async_b128(&As[buf][row * SA + cb + 8], gA + ko + 8);
    async_b128(&Bt[buf][row * SA + cb],     gB + ko);
    async_b128(&Bt[buf][row * SA + cb + 8], gB + ko + 8);
  };

  constexpr int NIT = D_ / 32;
  issue(0, 0);
  for (int it = 0; it < NIT; ++it) {
    wait_async0();
    __syncthreads();
    if (it + 1 < NIT) issue((it + 1) & 1, (it + 1) * 32);

    const _Float16* a_lds = As[it & 1];
    const _Float16* b_lds = Bt[it & 1];
    v16h af[4], bf[2];
#pragma unroll
    for (int ti = 0; ti < 4; ++ti) af[ti] = frag_a(a_lds, wm + ti * 16, SA, 0, lane);
#pragma unroll
    for (int tj = 0; tj < 2; ++tj) bf[tj] = frag_b(b_lds, wn + tj * 16, SA, 0, lane);
#pragma unroll
    for (int ti = 0; ti < 4; ++ti)
#pragma unroll
      for (int tj = 0; tj < 2; ++tj)
        acc[ti * 2 + tj] = wmma_f16(af[ti], bf[tj], acc[ti * 2 + tj]);
  }

#pragma unroll
  for (int ti = 0; ti < 4; ++ti) {
#pragma unroll
    for (int tj = 0; tj < 2; ++tj) {
      const int n  = gn0 + wn + tj * 16 + (lane & 15);
      const int mb = gm0 + wm + ti * 16 + ((lane >> 4) << 3);
      const float bias = bo[n];
#pragma unroll
      for (int i = 0; i < 8; ++i)
        out[(size_t)(mb + i) * D_ + n] = acc[ti * 2 + tj][i] + bias;
    }
  }
}

// ---------------------------------------------------------------------------
// Host launcher
// ---------------------------------------------------------------------------
extern "C" void kernel_launch(void* const* d_in, const int* in_sizes, int n_in,
                              void* d_out, int out_size, void* d_ws, size_t ws_size,
                              hipStream_t stream) {
  const float*   x    = (const float*)d_in[0];
  const uint8_t* mask = (const uint8_t*)d_in[1];
  const float*   t    = (const float*)d_in[2];
  const float*   p    = (const float*)d_in[3];
  const float*   Wq   = (const float*)d_in[4];
  const float*   bq   = (const float*)d_in[5];
  const float*   Wk   = (const float*)d_in[6];
  const float*   bk   = (const float*)d_in[7];
  const float*   Wv   = (const float*)d_in[8];
  const float*   bv   = (const float*)d_in[9];
  const float*   Wo   = (const float*)d_in[10];
  const float*   bo   = (const float*)d_in[11];

  float* out = (float*)d_out;                        // [B,L,D]
  float* w   = out + (size_t)B_ * L_ * D_;           // [B,H,L,L]

  char* ws = (char*)d_ws;
  size_t off = 0;
  auto alloc = [&](size_t bytes) -> void* {
    void* r = ws + off;
    off += (bytes + 255) & ~(size_t)255;
    return r;
  };
  _Float16* x16   = (_Float16*)alloc((size_t)ML_ * D_ * 2);
  _Float16* WqT   = (_Float16*)alloc((size_t)D_ * D_ * 2);
  _Float16* WkT   = (_Float16*)alloc((size_t)D_ * D_ * 2);
  _Float16* WvT   = (_Float16*)alloc((size_t)D_ * D_ * 2);
  _Float16* WoT   = (_Float16*)alloc((size_t)D_ * D_ * 2);
  _Float16* q16   = (_Float16*)alloc((size_t)BH_ * L_ * DH_ * 2);
  _Float16* k16   = (_Float16*)alloc((size_t)BH_ * L_ * DH_ * 2);
  _Float16* v16t  = (_Float16*)alloc((size_t)BH_ * L_ * DH_ * 2);
  float*    biasv = (float*)alloc((size_t)B_ * L_ * 4);
  _Float16* ao16  = (_Float16*)alloc((size_t)ML_ * D_ * 2);

  k_cvt<<<2048, 256, 0, stream>>>(x, x16, ML_ * D_);
  k_wtrans<<<dim3(32, 32, 4), dim3(32, 8), 0, stream>>>(Wq, Wk, Wv, Wo,
                                                        WqT, WkT, WvT, WoT);
  k_qkv_gemm<<<dim3(D_ / 128, ML_ / 128, 3), 256, 0, stream>>>(
      x16, WqT, WkT, WvT, bq, bk, bv, q16, k16, v16t);
  k_bias<<<B_ * L_, 256, 0, stream>>>(t, p, biasv);
  k_scores<<<dim3(L_ / 128, L_ / 128, BH_), 256, 0, stream>>>(q16, k16, biasv,
                                                              mask, w);
  k_softmax<<<BH_ * L_, 256, 0, stream>>>(w);
  k_attnout<<<dim3(L_ / 128, 1, BH_), 256, 0, stream>>>(w, v16t, ao16);
  k_outproj<<<dim3(D_ / 128, ML_ / 128, 1), 256, 0, stream>>>(ao16, WoT, bo, out);
}
